// MessagePassingUnitGatingWithRelnessLogits_31181462569557
// MI455X (gfx1250) — matrix operand order repaired and checked by reading, not compiled, not run on hardware
//
#include <hip/hip_runtime.h>
#include <hip/hip_bf16.h>

typedef __attribute__((ext_vector_type(16))) __bf16 v16bf;
typedef __attribute__((ext_vector_type(8)))  float  v8f;

#define D_DIM      1024
#define TWO_D      2048
#define A_DIM      256
#define FD_DIM     64
#define CAT_STRIDE 2056   // bf16 elems per row -> 1028 dwords/row (even: b64-aligned stores)
#define CAT_DW     1028
#define AUX_STRIDE 260    // f32 elems per row; breaks 4-way bank conflict across rows
#define LN_EPS     1e-5f

__device__ __forceinline__ unsigned short f32_to_bf16_rne(float f) {
    unsigned int u = __float_as_uint(f);
    unsigned int r = u + 0x7FFFu + ((u >> 16) & 1u);
    return (unsigned short)(r >> 16);
}
__device__ __forceinline__ unsigned int pack_bf16x2(float lo, float hi) {
    return (unsigned int)f32_to_bf16_rne(lo) | ((unsigned int)f32_to_bf16_rne(hi) << 16);
}
__device__ __forceinline__ float bf16_lo(unsigned int w) {
    return __uint_as_float(w << 16);
}
__device__ __forceinline__ float bf16_hi(unsigned int w) {
    return __uint_as_float(w & 0xFFFF0000u);
}
__device__ __forceinline__ float sigmoidf_(float x) {
    return __builtin_amdgcn_rcpf(1.0f + __expf(-x));   // v_rcp_f32, no IEEE div chain
}

// ---------------------------------------------------------------------------
// Pre-kernel: pack W [2048,64] f32 (row-major) into bf16 B-fragments for
// v_wmma_f32_16x16x32_bf16.  Layout (contiguous k = 16h + s order, same as A):
//   element (K, n):  lane = (n%16) + 16*((K%32)/16), vgpr = (K%16)/2, half = K%2
// Linear index: ((t*64 + s)*32 + lane)*8 + v   (t = n/16 tile, s = K/32 step)
// ---------------------------------------------------------------------------
__global__ void pack_w_kernel(const float* __restrict__ W,
                              unsigned int* __restrict__ wpack) {
    int tid = blockIdx.x * blockDim.x + threadIdx.x;   // 0 .. 65535
    if (tid >= 4 * 64 * 32 * 8) return;
    int v = tid & 7;
    int l = (tid >> 3) & 31;
    int s = (tid >> 8) & 63;
    int t = tid >> 14;
    int n   = 16 * t + (l & 15);
    int kk0 = 16 * (l >> 4) + 2 * v;
    int K   = 32 * s + kk0;
    wpack[tid] = pack_bf16x2(W[(size_t)K * FD_DIM + n], W[(size_t)(K + 1) * FD_DIM + n]);
}

// ---------------------------------------------------------------------------
// Fused kernel: one block = 16 rows, 256 threads = 8 waves.
// ---------------------------------------------------------------------------
__global__ __launch_bounds__(256)
void mpu_gating_kernel(const float* __restrict__ unary,
                       const float* __restrict__ pair,
                       const float* __restrict__ aux,
                       const float* __restrict__ auxw,
                       const float* __restrict__ lng,
                       const float* __restrict__ lnb,
                       const unsigned int* __restrict__ wpack,
                       const float* __restrict__ bias,
                       const float* __restrict__ lnag,
                       const float* __restrict__ lnab,
                       const float* __restrict__ Waux,
                       const float* __restrict__ baux,
                       const float* __restrict__ gwp,
                       const float* __restrict__ agwp,
                       float* __restrict__ out,
                       float* __restrict__ out_g) {
    __shared__ unsigned short s_cat[16 * CAT_STRIDE];   // relu(cat) -> normalized, bf16
    __shared__ float s_aux[16 * AUX_STRIDE];            // aux normalized, f32
    __shared__ float s_mu[16], s_inv[16];
    __shared__ float s_gate[16];
    __shared__ float s_auxg[16];
    __shared__ float s_g[16];

    const int tid  = threadIdx.x;
    const int lane = tid & 31;
    const int wave = tid >> 5;
    const long long row0 = (long long)blockIdx.x * 16;

    if (tid < 16) { s_gate[tid] = 0.0f; s_auxg[tid] = 0.0f; }

    // ---- P1: stats over relu(cat([unary,pair])), stage bf16 in LDS ----------
    for (int rr = 0; rr < 2; ++rr) {
        const int r = 2 * wave + rr;
        const float4* __restrict__ su4 = (const float4*)(unary + (row0 + r) * D_DIM);
        const float4* __restrict__ sp4 = (const float4*)(pair  + (row0 + r) * D_DIM);
        uint2* __restrict__ catrow = (uint2*)((unsigned int*)s_cat + r * CAT_DW);
        float s = 0.0f, ss = 0.0f;
        #pragma unroll 4
        for (int it = 0; it < 16; ++it) {
            int q = it * 32 + lane;                       // float4 index 0..511
            float4 x = (q < 256) ? su4[q] : sp4[q - 256];
            float a0 = fmaxf(x.x, 0.0f), a1 = fmaxf(x.y, 0.0f);
            float a2 = fmaxf(x.z, 0.0f), a3 = fmaxf(x.w, 0.0f);
            s  += (a0 + a1) + (a2 + a3);
            ss += (a0 * a0 + a1 * a1) + (a2 * a2 + a3 * a3);
            uint2 p; p.x = pack_bf16x2(a0, a1); p.y = pack_bf16x2(a2, a3);
            catrow[q] = p;                                // b64 LDS store
        }
        #pragma unroll
        for (int m = 1; m < 32; m <<= 1) {
            s  += __shfl_xor(s, m, 32);
            ss += __shfl_xor(ss, m, 32);
        }
        if (lane == 0) {
            float mu  = s * (1.0f / 2048.0f);
            float var = ss * (1.0f / 2048.0f) - mu * mu;
            s_mu[r]  = mu;
            s_inv[r] = rsqrtf(var + LN_EPS);
        }
    }
    __syncthreads();

    // ---- P1.5: LayerNorm + relu in place (dword = 2 bf16 per iter) ----------
    {
        unsigned int* cat32 = (unsigned int*)s_cat;
        const float2* __restrict__ lng2 = (const float2*)lng;
        const float2* __restrict__ lnb2 = (const float2*)lnb;
        #pragma unroll 4
        for (int i = 0; i < 64; ++i) {
            int u  = tid + i * 256;        // dword index 0..16383
            int r  = u >> 10;
            int kd = u & 1023;
            unsigned int w = cat32[r * CAT_DW + kd];
            float2 g = lng2[kd];
            float2 b = lnb2[kd];
            float mu = s_mu[r], inv = s_inv[r];
            float y0 = fmaxf((bf16_lo(w) - mu) * inv * g.x + b.x, 0.0f);
            float y1 = fmaxf((bf16_hi(w) - mu) * inv * g.y + b.y, 0.0f);
            cat32[r * CAT_DW + kd] = pack_bf16x2(y0, y1);
        }
    }
    __syncthreads();

    // ---- P2: waves 0-3 WMMA gate matmul, waves 4-7 aux branch ---------------
    if (wave < 4) {
        const int t = wave;                      // output-column tile (16 cols)
        const int m_row = lane & 15;
        const int h     = lane >> 4;
        const unsigned int* __restrict__ cat32 = (const unsigned int*)s_cat;
        const unsigned int* __restrict__ wp = wpack + (((size_t)t * 64) * 32 + lane) * 8;

        int dwBase[8];
        #pragma unroll
        for (int v = 0; v < 8; ++v) {
            // A-fragment (contiguous k = 16h + s order, same as wpack): lane half h picks K group
            int kk0 = 16 * h + 2 * v;
            dwBase[v] = m_row * CAT_DW + (kk0 >> 1);
        }

        v8f acc = {};
        for (int s = 0; s < 64; ++s) {
            union { unsigned int u[8]; v16bf v; } ua, ub;
            #pragma unroll
            for (int v = 0; v < 8; ++v) ua.u[v] = cat32[dwBase[v] + s * 16];
            #pragma unroll
            for (int v = 0; v < 8; ++v) ub.u[v] = wp[(size_t)s * 256 + v];
            acc = __builtin_amdgcn_wmma_f32_16x16x32_bf16(
                false, ua.v, false, ub.v, (short)0, acc, false, false);
        }

        const float bn = bias[t * 16 + (lane & 15)];
        // element (M = v + 8h, n = 16t + lane%16): every lane adds its sigmoid
        // directly into the per-row LDS accumulator (ds_add_f32, no shuffles).
        #pragma unroll
        for (int v = 0; v < 8; ++v) {
            atomicAdd(&s_gate[v + 8 * h], sigmoidf_(acc[v] + bn));
        }
    } else {
        // aux branch: wave-4 handles 4 rows
        const int wbase = 4 * (wave - 4);
        for (int rr = 0; rr < 4; ++rr) {
            const int r = wbase + rr;
            const float* __restrict__ src = aux + (row0 + r) * A_DIM;
            float s = 0.0f, ss = 0.0f;
            float xv[8];
            #pragma unroll
            for (int j = 0; j < 8; ++j) {
                float x = src[j * 32 + lane];
                xv[j] = x; s += x; ss += x * x;
            }
            #pragma unroll
            for (int m = 1; m < 32; m <<= 1) {
                s  += __shfl_xor(s, m, 32);
                ss += __shfl_xor(ss, m, 32);
            }
            float mu  = s * (1.0f / 256.0f);
            float var = ss * (1.0f / 256.0f) - mu * mu;
            float inv = rsqrtf(var + LN_EPS);
            #pragma unroll
            for (int j = 0; j < 8; ++j) {
                int k = j * 32 + lane;
                float y = fmaxf((xv[j] - mu) * inv * lnag[k] + lnab[k], 0.0f);
                s_aux[r * AUX_STRIDE + k] = y;
            }
        }
        // 256x8 matmul: lane -> (row = lane/8, col = lane%8)
        const int r   = wbase + (lane >> 3);
        const int col = lane & 7;
        float acc = 0.0f;
        for (int k = 0; k < 256; ++k)
            acc += s_aux[r * AUX_STRIDE + k] * Waux[k * 8 + col];
        atomicAdd(&s_auxg[r], sigmoidf_(acc + baux[col]));
    }
    __syncthreads();

    // ---- P3: reverse-sigmoid combine --------------------------------------
    if (wave == 0 && lane < 16) {
        const float gw  = gwp[0];
        const float agw = agwp[0];
        float gate = s_gate[lane] * (1.0f / 64.0f);
        float ag   = s_auxg[lane] * (1.0f / 8.0f);
        float y1 = fminf(fmaxf(gate, 0.001f), 0.999f);
        float y2 = fminf(fmaxf(ag,   0.001f), 0.999f);
        float l1 = __logf(y1) - __logf(1.0f - y1);
        float l2 = __logf(y2) - __logf(1.0f - y2);
        float g  = sigmoidf_(gw * l1 + agw * l2) * auxw[row0 + lane];
        out_g[row0 + lane] = g;
        s_g[lane] = g;
    }
    __syncthreads();

    // ---- P4: output = pair * g (pair rows are L2-hot from P1) ---------------
    const long long base = row0 * (long long)D_DIM;
    const float4* __restrict__ pair4 = (const float4*)(pair + base);
    float4* __restrict__ out4 = (float4*)(out + base);
    #pragma unroll 4
    for (int i = 0; i < 16; ++i) {
        int e = tid + i * 256;             // float4 index 0..4095
        float g = s_g[e >> 8];
        float4 p = pair4[e];
        p.x *= g; p.y *= g; p.z *= g; p.w *= g;
        out4[e] = p;
    }
}

extern "C" void kernel_launch(void* const* d_in, const int* in_sizes, int n_in,
                              void* d_out, int out_size, void* d_ws, size_t ws_size,
                              hipStream_t stream) {
    const float* unary = (const float*)d_in[0];
    const float* pair  = (const float*)d_in[1];
    const float* aux   = (const float*)d_in[2];
    const float* auxw  = (const float*)d_in[3];
    const float* lng   = (const float*)d_in[4];
    const float* lnb   = (const float*)d_in[5];
    const float* W     = (const float*)d_in[6];
    const float* bias  = (const float*)d_in[7];
    const float* lnag  = (const float*)d_in[8];
    const float* lnab  = (const float*)d_in[9];
    const float* Waux  = (const float*)d_in[10];
    const float* baux  = (const float*)d_in[11];
    const float* gwp   = (const float*)d_in[12];
    const float* agwp  = (const float*)d_in[13];

    const int N = in_sizes[3];                    // 32768
    float* out   = (float*)d_out;
    float* out_g = out + (long long)N * D_DIM;
    unsigned int* wpack = (unsigned int*)d_ws;    // 4*64*32*8 dwords = 256 KB

    pack_w_kernel<<<256, 256, 0, stream>>>(W, wpack);
    mpu_gating_kernel<<<N / 16, 256, 0, stream>>>(
        unary, pair, aux, auxw, lng, lnb, wpack, bias,
        lnag, lnab, Waux, baux, gwp, agwp, out, out_g);
}
